// MTGNN_5978594476533
// MI455X (gfx1250) — matrix operand order repaired and checked by reading, not compiled, not run on hardware
//
#include <hip/hip_runtime.h>
#include <hip/hip_bf16.h>
#include <math.h>

// ---------------------------------------------------------------------------
// MTGNN forward for MI455X (gfx1250). Dominant cost: mixprop propagation
// GEMMs [M x 2048] @ [2048 x 2048] done with v_wmma_f32_16x16x32_bf16,
// fed by double-buffered GLOBAL_LOAD_ASYNC_TO_LDS_B128 staging (ASYNCcnt).
// Layout convention for pipeline tensors: [b][c][t][n], n contiguous (2048).
// ---------------------------------------------------------------------------

typedef __attribute__((ext_vector_type(8)))  float  v8f;
typedef __attribute__((ext_vector_type(16))) __bf16 v16bf;
typedef __attribute__((ext_vector_type(8)))  __bf16 v8bf;

static __device__ inline __bf16 f2bf(float f) {
    unsigned u = __builtin_bit_cast(unsigned, f);
    unsigned short h;
    if (f != f) {
        h = (unsigned short)((u >> 16) | 0x40u);      // quiet the NaN
    } else {
        unsigned r = u + 0x7fffu + ((u >> 16) & 1u);  // round-to-nearest-even
        h = (unsigned short)(r >> 16);
    }
    return __builtin_bit_cast(__bf16, h);
}

// Low 32 bits of a generic pointer to __shared__ data are the LDS offset
// (ISA 10.2: flat->LDS uses addr[31:0]; addrspace(3) pointers are offsets).
static __device__ inline unsigned lds_off(const void* p) {
    return (unsigned)(unsigned long long)p;
}

// ---------------------------------------------------------------------------
// 1) Graph constructor: n1 = tanh(3*(emb1@lin1_w.T + b1)), n2 likewise
// ---------------------------------------------------------------------------
__global__ __launch_bounds__(256) void k_embed(
    const float* __restrict__ emb1, const float* __restrict__ emb2,
    const float* __restrict__ l1w, const float* __restrict__ l1b,
    const float* __restrict__ l2w, const float* __restrict__ l2b,
    float* __restrict__ n1, float* __restrict__ n2) {
    int idx = blockIdx.x * 256 + threadIdx.x;
    if (idx >= 2 * 2048 * 40) return;
    int o = idx % 40;
    int r = (idx / 40) % 2048;
    int which = idx / (2048 * 40);
    const float* e = which ? emb2 : emb1;
    const float* w = which ? l2w : l1w;
    const float* bb = which ? l2b : l1b;
    float s = bb[o];
    #pragma unroll
    for (int j = 0; j < 40; ++j) s += e[r * 40 + j] * w[o * 40 + j];
    (which ? n2 : n1)[r * 40 + o] = tanhf(3.0f * s);
}

// adj0 = relu(tanh(3*(n1@n2.T - n2@n1.T)))
__global__ __launch_bounds__(256) void k_adj(
    const float* __restrict__ n1, const float* __restrict__ n2,
    float* __restrict__ adj0) {
    __shared__ float A1[16][40], A2[16][40], B1[16][40], B2[16][40];
    int r0 = blockIdx.y * 16, c0 = blockIdx.x * 16;
    int tid = threadIdx.x;
    for (int i = tid; i < 16 * 40; i += 256) {
        int rr = i / 40, jj = i % 40;
        A1[rr][jj] = n1[(size_t)(r0 + rr) * 40 + jj];
        A2[rr][jj] = n2[(size_t)(r0 + rr) * 40 + jj];
        B1[rr][jj] = n1[(size_t)(c0 + rr) * 40 + jj];
        B2[rr][jj] = n2[(size_t)(c0 + rr) * 40 + jj];
    }
    __syncthreads();
    int ty = tid >> 4, tx = tid & 15;
    float d1 = 0.f, d2 = 0.f;
    #pragma unroll
    for (int j = 0; j < 40; ++j) {
        d1 += A1[ty][j] * B2[tx][j];
        d2 += A2[ty][j] * B1[tx][j];
    }
    float a = tanhf(3.0f * (d1 - d2));
    adj0[(size_t)(r0 + ty) * 2048 + (c0 + tx)] = a > 0.f ? a : 0.f;
}

__global__ __launch_bounds__(256) void k_fill(float* __restrict__ p, float v, int count) {
    int idx = blockIdx.x * 256 + threadIdx.x;
    if (idx < count) p[idx] = v;
}

// Per-row top-20: scatter values into adp, emit rowsum r1 = 1 + sum(selected)
__global__ __launch_bounds__(256) void k_topk(
    const float* __restrict__ adj0, float* __restrict__ adp, float* __restrict__ r1) {
    __shared__ float vals[2048];
    __shared__ float rv[256];
    __shared__ int   ri[256];
    int row = blockIdx.x;
    int tid = threadIdx.x;
    const float* src = adj0 + (size_t)row * 2048;
    for (int i = tid; i < 2048; i += 256) vals[i] = src[i];
    __syncthreads();
    float rsum = 0.f;
    for (int it = 0; it < 20; ++it) {
        float lv = -1e30f; int li = tid * 8;
        #pragma unroll
        for (int e = 0; e < 8; ++e) {
            float v = vals[tid * 8 + e];
            if (v > lv) { lv = v; li = tid * 8 + e; }
        }
        rv[tid] = lv; ri[tid] = li;
        __syncthreads();
        for (int s = 128; s > 0; s >>= 1) {
            if (tid < s) {
                if (rv[tid + s] > rv[tid] ||
                    (rv[tid + s] == rv[tid] && ri[tid + s] < ri[tid])) {
                    rv[tid] = rv[tid + s]; ri[tid] = ri[tid + s];
                }
            }
            __syncthreads();
        }
        if (tid == 0) {
            int bi = ri[0];
            float bv = rv[0];
            adp[(size_t)row * 2048 + bi] = bv;
            rsum += bv;
            vals[bi] = -1e30f;
        }
        __syncthreads();
    }
    if (tid == 0) r1[row] = 1.0f + rsum;
}

// r2[n] = 1 + column-sum of adp
__global__ __launch_bounds__(256) void k_colsum(
    const float* __restrict__ adp, float* __restrict__ r2) {
    int n = blockIdx.x * 256 + threadIdx.x;
    if (n >= 2048) return;
    float s = 0.f;
    for (int v = 0; v < 2048; ++v) s += adp[(size_t)v * 2048 + n];
    r2[n] = 1.0f + s;
}

// Bt1[n][k] = (adp[k][n] + (k==n)) / r1[k]   (An1 transposed, bf16)
// Bt2[n][k] = (adp[n][k] + (k==n)) / r2[k]   (An2 transposed, bf16)
__global__ __launch_bounds__(1024) void k_buildBt(
    const float* __restrict__ adp, const float* __restrict__ r1,
    const float* __restrict__ r2, __bf16* __restrict__ Bt1,
    __bf16* __restrict__ Bt2) {
    __shared__ float s[32][33];
    int k0 = blockIdx.x * 32, n0 = blockIdx.y * 32;
    int tx = threadIdx.x, ty = threadIdx.y;
    s[ty][tx] = adp[(size_t)(k0 + ty) * 2048 + (n0 + tx)];
    __syncthreads();
    {   // Bt1: n = n0+ty, k = k0+tx
        int n = n0 + ty, k = k0 + tx;
        float v = (s[tx][ty] + (k == n ? 1.f : 0.f)) / r1[k];
        Bt1[(size_t)n * 2048 + k] = f2bf(v);
    }
    {   // Bt2: n = k0+ty, k = n0+tx
        int n = k0 + ty, k = n0 + tx;
        float v = (s[ty][tx] + (n == k ? 1.f : 0.f)) / r2[k];
        Bt2[(size_t)n * 2048 + k] = f2bf(v);
    }
}

// ---------------------------------------------------------------------------
// 2) start conv (1x1, pad time front to RF=19) and skip0 (1x19 conv -> T=1)
// ---------------------------------------------------------------------------
__global__ __launch_bounds__(256) void k_start(
    const float* __restrict__ x, const float* __restrict__ w,
    const float* __restrict__ b, float* __restrict__ h) {
    int idx = blockIdx.x * 256 + threadIdx.x;              // [16][32][19][2048]
    if (idx >= 16 * 32 * 19 * 2048) return;
    int n = idx & 2047;
    int r = idx >> 11;
    int t = r % 19; r /= 19;
    int c = r & 31; int bb = r >> 5;
    float acc = b[c];
    if (t >= 7) {
        #pragma unroll
        for (int ci = 0; ci < 2; ++ci)
            acc += w[c * 2 + ci] * x[((size_t)(bb * 2 + ci) * 2048 + n) * 12 + (t - 7)];
    }
    h[idx] = acc;
}

__global__ __launch_bounds__(256) void k_skip0(
    const float* __restrict__ x, const float* __restrict__ w,
    const float* __restrict__ b, float* __restrict__ skip) {
    int idx = blockIdx.x * 256 + threadIdx.x;              // [16][64][2048]
    if (idx >= 16 * 64 * 2048) return;
    int n = idx & 2047;
    int r = idx >> 11;
    int co = r & 63; int bb = r >> 6;
    float acc = b[co];
    #pragma unroll
    for (int ci = 0; ci < 2; ++ci)
        for (int t = 0; t < 12; ++t)
            acc += w[co * 38 + ci * 19 + (7 + t)] *
                   x[((size_t)(bb * 2 + ci) * 2048 + n) * 12 + t];
    skip[idx] = acc;
}

// ---------------------------------------------------------------------------
// 3) dilated inception: hg = tanh(filter) * sigmoid(gate)
// ---------------------------------------------------------------------------
__global__ __launch_bounds__(256) void k_incept(
    const float* __restrict__ h, float* __restrict__ hg,
    const float* __restrict__ fw2, const float* __restrict__ fw3,
    const float* __restrict__ fw6, const float* __restrict__ fw7,
    const float* __restrict__ fb,
    const float* __restrict__ gw2, const float* __restrict__ gw3,
    const float* __restrict__ gw6, const float* __restrict__ gw7,
    const float* __restrict__ gb, int T, int Tin) {
    int idx = blockIdx.x * 256 + threadIdx.x;
    if (idx >= 16 * 32 * T * 2048) return;
    int n = idx & 2047;
    int r = idx >> 11;
    int t = r % T; r /= T;
    int c = r & 31; int bb = r >> 5;
    int br = c >> 3, oc = c & 7;
    int k = (br == 0) ? 2 : (br == 1) ? 3 : (br == 2) ? 6 : 7;
    const float* fw = (br == 0) ? fw2 : (br == 1) ? fw3 : (br == 2) ? fw6 : fw7;
    const float* gw = (br == 0) ? gw2 : (br == 1) ? gw3 : (br == 2) ? gw6 : gw7;
    float fs = fb[br * 8 + oc], gs = gb[br * 8 + oc];
    for (int ci = 0; ci < 32; ++ci) {
        const float* hp = h + ((size_t)(bb * 32 + ci) * Tin + (t + 7 - k)) * 2048 + n;
        const float* fwp = fw + (oc * 32 + ci) * k;
        const float* gwp = gw + (oc * 32 + ci) * k;
        for (int j = 0; j < k; ++j) {
            float xv = hp[(size_t)j * 2048];
            fs += fwp[j] * xv;
            gs += gwp[j] * xv;
        }
    }
    float sg = 1.0f / (1.0f + expf(-gs));
    hg[((size_t)(bb * 32 + c) * T + t) * 2048 + n] = tanhf(fs) * sg;
}

// skip conv (kernel width == T -> output T=1), accumulates into skip
__global__ __launch_bounds__(256) void k_skipconv(
    const float* __restrict__ hg, float* __restrict__ skip,
    const float* __restrict__ w, const float* __restrict__ b, int T) {
    int idx = blockIdx.x * 256 + threadIdx.x;              // [16][64][2048]
    if (idx >= 16 * 64 * 2048) return;
    int n = idx & 2047;
    int r = idx >> 11;
    int co = r & 63; int bb = r >> 6;
    float acc = b[co];
    for (int ci = 0; ci < 32; ++ci)
        for (int t = 0; t < T; ++t)
            acc += w[(co * 32 + ci) * T + t] *
                   hg[((size_t)(bb * 32 + ci) * T + t) * 2048 + n];
    skip[idx] += acc;
}

// ---------------------------------------------------------------------------
// 4) bf16 conversion + WMMA mixprop GEMM, async-LDS staged, fused epilogue
//    D = alpha*H + (1-alpha) * (G @ An),  An fed transposed (Bt[n][k], bf16)
// ---------------------------------------------------------------------------
__global__ __launch_bounds__(256) void k_tobf(
    const float* __restrict__ src, __bf16* __restrict__ dst, int count) {
    int idx = blockIdx.x * 256 + threadIdx.x;
    if (idx < count) dst[idx] = f2bf(src[idx]);
}

#define LDSROW 40   // 32 bf16 payload + 8 bf16 pad (80B rows, bank-friendly)

__global__ __launch_bounds__(256) void k_gemm(
    const __bf16* __restrict__ G,   // [M x 2048] bf16 row-major
    const __bf16* __restrict__ Bt,  // [2048 x 2048] bf16, Bt[n][k] = An[k][n]
    const float*  __restrict__ H,   // [M x 2048] f32 (blend source)
    float*        __restrict__ D,   // [M x 2048] f32 out
    __bf16*       __restrict__ Dbf, // optional bf16 copy of D
    int M, float alpha) {
    // Block tile: 64 (M) x 256 (N). 8 waves = 2 (M) x 4 (N); wave tile 32x64.
    __shared__ __bf16 Abuf[2][64 * LDSROW];    //  5 KB x2
    __shared__ __bf16 Bbuf[2][256 * LDSROW];   // 20 KB x2
    const int lane = threadIdx.x & 31;
    const int wave = threadIdx.x >> 5;
    const int wr = wave >> 2;                  // 0..1  (M wave-row)
    const int wc = wave & 3;                   // 0..3  (N wave-col)
    const int mblk = blockIdx.y * 64;
    const int nblk = blockIdx.x * 256;
    const int mr = lane & 15;
    const int kh = lane >> 4;
    (void)M;

    // --- async stage issue: 5 x b128 per wave per stage (4 B-segs + 1 A-seg)
    auto issue_stage = [&](int k0, int buf) {
        #pragma unroll
        for (int i2 = 0; i2 < 4; ++i2) {
            int seg = i2 * 256 + wave * 32 + lane;     // 1024 segs = 256 rows x 4
            int row = seg >> 2, ch = seg & 3;
            const __bf16* gp = Bt + (size_t)(nblk + row) * 2048 + k0 + ch * 8;
            unsigned lo = lds_off(&Bbuf[buf][row * LDSROW + ch * 8]);
#if defined(__gfx1250__)
            asm volatile("global_load_async_to_lds_b128 %0, %1, off"
                         :: "v"(lo), "v"(gp) : "memory");
#else
            (void)gp; (void)lo;
#endif
        }
        {
            int seg = wave * 32 + lane;                // 256 segs = 64 rows x 4
            int row = seg >> 2, ch = seg & 3;
            const __bf16* gp = G + (size_t)(mblk + row) * 2048 + k0 + ch * 8;
            unsigned lo = lds_off(&Abuf[buf][row * LDSROW + ch * 8]);
#if defined(__gfx1250__)
            asm volatile("global_load_async_to_lds_b128 %0, %1, off"
                         :: "v"(lo), "v"(gp) : "memory");
#else
            (void)gp; (void)lo;
#endif
        }
    };

    v8f acc[2][4];
    #pragma unroll
    for (int mi = 0; mi < 2; ++mi)
        #pragma unroll
        for (int j = 0; j < 4; ++j) acc[mi][j] = (v8f){};

    issue_stage(0, 0);
    for (int s = 0; s < 64; ++s) {
        const int cur = s & 1;
        if (s + 1 < 64) {
            issue_stage((s + 1) * 32, cur ^ 1);
#if defined(__gfx1250__)
            asm volatile("s_wait_asynccnt 0x5" ::: "memory");  // stage s done
#endif
        } else {
#if defined(__gfx1250__)
            asm volatile("s_wait_asynccnt 0x0" ::: "memory");
#endif
        }
        __syncthreads();

        // A fragments (two 16-row tiles of this wave's 32 rows)
        v16bf a0, a1;
        {
            const __bf16* ap = &Abuf[cur][(wr * 32 + mr) * LDSROW + 8 * kh];
            v8bf lo = *(const v8bf*)ap;
            v8bf hi = *(const v8bf*)(ap + 16);
            #pragma unroll
            for (int e = 0; e < 8; ++e) { a0[e] = lo[e]; a0[e + 8] = hi[e]; }
        }
        {
            const __bf16* ap = &Abuf[cur][(wr * 32 + 16 + mr) * LDSROW + 8 * kh];
            v8bf lo = *(const v8bf*)ap;
            v8bf hi = *(const v8bf*)(ap + 16);
            #pragma unroll
            for (int e = 0; e < 8; ++e) { a1[e] = lo[e]; a1[e + 8] = hi[e]; }
        }
        #pragma unroll
        for (int j = 0; j < 4; ++j) {
            const __bf16* bp = &Bbuf[cur][(wc * 64 + j * 16 + mr) * LDSROW + 8 * kh];
            v8bf l0 = *(const v8bf*)bp;
            v8bf l1 = *(const v8bf*)(bp + 16);
            v16bf bfr;
            #pragma unroll
            for (int e = 0; e < 8; ++e) { bfr[e] = l0[e]; bfr[e + 8] = l1[e]; }
#if defined(__gfx1250__)
            acc[0][j] = __builtin_amdgcn_wmma_f32_16x16x32_bf16(
                false, a0, false, bfr, (short)0, acc[0][j], false, false);
            acc[1][j] = __builtin_amdgcn_wmma_f32_16x16x32_bf16(
                false, a1, false, bfr, (short)0, acc[1][j], false, false);
#else
            (void)a0; (void)a1; (void)bfr;
#endif
        }
        __syncthreads();   // protect buffer 'cur^1'... (reuse two stages later)
    }

    const float beta = 1.0f - alpha;
    #pragma unroll
    for (int mi = 0; mi < 2; ++mi) {
        #pragma unroll
        for (int j = 0; j < 4; ++j) {
            #pragma unroll
            for (int r = 0; r < 8; ++r) {
                int m = mblk + wr * 32 + mi * 16 + r + 8 * kh;
                int n = nblk + wc * 64 + 16 * j + mr;
                size_t off = (size_t)m * 2048 + n;
                float v = alpha * H[off] + beta * acc[mi][j][r];
                D[off] = v;
                if (Dbf) Dbf[off] = f2bf(v);
            }
        }
    }
}

// 1x1 conv over concat([h,g1,g2]) -> 32 ch; optional accumulate + residual
__global__ __launch_bounds__(256) void k_mixconv(
    const float* __restrict__ h, const float* __restrict__ g1,
    const float* __restrict__ g2, const float* __restrict__ w,
    const float* __restrict__ bias, float* __restrict__ out,
    int T, int accFlag, const float* __restrict__ res, int Tres) {
    int idx = blockIdx.x * 256 + threadIdx.x;              // (b,t,n) pixels
    if (idx >= 16 * T * 2048) return;
    int n = idx & 2047;
    int bt = idx >> 11;
    int t = bt % T, bb = bt / T;
    float acc[32];
    #pragma unroll
    for (int co = 0; co < 32; ++co) acc[co] = bias[co];
    for (int cc = 0; cc < 32; ++cc) {
        size_t off = ((size_t)(bb * 32 + cc) * T + t) * 2048 + n;
        float xh = h[off], x1 = g1[off], x2 = g2[off];
        #pragma unroll
        for (int co = 0; co < 32; ++co)
            acc[co] += w[co * 96 + cc] * xh + w[co * 96 + 32 + cc] * x1 +
                       w[co * 96 + 64 + cc] * x2;
    }
    #pragma unroll
    for (int co = 0; co < 32; ++co) {
        size_t o = ((size_t)(bb * 32 + co) * T + t) * 2048 + n;
        float v = acc[co];
        if (accFlag) v += out[o];
        if (res) v += res[((size_t)(bb * 32 + co) * Tres + (t + 6)) * 2048 + n];
        out[o] = v;
    }
}

// LayerNorm over (C,N,T) per batch element; weights indexed [c][n][t]
__global__ __launch_bounds__(256) void k_lnorm(
    const float* __restrict__ x, float* __restrict__ y,
    const float* __restrict__ w, const float* __restrict__ bias, int T) {
    __shared__ float rs[256], rss[256];
    __shared__ float stat[2];
    int bb = blockIdx.x, tid = threadIdx.x;
    size_t base = (size_t)bb * 32 * T * 2048;
    int count = 32 * T * 2048;
    float s = 0.f, ss = 0.f;
    for (int i = tid; i < count; i += 256) {
        float v = x[base + i];
        s += v; ss += v * v;
    }
    rs[tid] = s; rss[tid] = ss;
    __syncthreads();
    for (int st = 128; st > 0; st >>= 1) {
        if (tid < st) { rs[tid] += rs[tid + st]; rss[tid] += rss[tid + st]; }
        __syncthreads();
    }
    if (tid == 0) {
        float mu = rs[0] / count;
        float var = rss[0] / count - mu * mu;
        stat[0] = mu;
        stat[1] = rsqrtf(var + 1e-5f);
    }
    __syncthreads();
    float mu = stat[0], rsig = stat[1];
    for (int i = tid; i < count; i += 256) {
        int n = i & 2047;
        int ct = i >> 11;
        int t = ct % T;
        int c = ct / T;
        size_t widx = (size_t)c * 2048 * T + (size_t)n * T + t;
        y[base + i] = (x[base + i] - mu) * rsig * w[widx] + bias[widx];
    }
}

// ---------------------------------------------------------------------------
// 5) output head
// ---------------------------------------------------------------------------
__global__ __launch_bounds__(256) void k_skipE(
    const float* __restrict__ skip, const float* __restrict__ h,
    const float* __restrict__ w, const float* __restrict__ b,
    float* __restrict__ sbuf) {
    int idx = blockIdx.x * 256 + threadIdx.x;              // [16][64][2048]
    if (idx >= 16 * 64 * 2048) return;
    int n = idx & 2047;
    int r = idx >> 11;
    int co = r & 63; int bb = r >> 6;
    float acc = skip[idx] + b[co];
    #pragma unroll
    for (int ci = 0; ci < 32; ++ci)
        acc += w[co * 32 + ci] * h[(size_t)(bb * 32 + ci) * 2048 + n];
    sbuf[idx] = acc > 0.f ? acc : 0.f;                     // relu(skip total)
}

__global__ __launch_bounds__(256) void k_end1(
    const float* __restrict__ s, const float* __restrict__ w,
    const float* __restrict__ b, float* __restrict__ e1) {
    int idx = blockIdx.x * 256 + threadIdx.x;              // [16][128][2048]
    if (idx >= 16 * 128 * 2048) return;
    int n = idx & 2047;
    int r = idx >> 11;
    int co = r & 127; int bb = r >> 7;
    float acc = b[co];
    #pragma unroll 8
    for (int ci = 0; ci < 64; ++ci)
        acc += w[co * 64 + ci] * s[(size_t)(bb * 64 + ci) * 2048 + n];
    e1[idx] = acc > 0.f ? acc : 0.f;
}

__global__ __launch_bounds__(256) void k_end2(
    const float* __restrict__ e1, const float* __restrict__ w,
    const float* __restrict__ b, float* __restrict__ out) {
    int idx = blockIdx.x * 256 + threadIdx.x;              // [16][12][2048]
    if (idx >= 16 * 12 * 2048) return;
    int n = idx & 2047;
    int r = idx >> 11;
    int od = r % 12; int bb = r / 12;
    float acc = b[od];
    #pragma unroll 8
    for (int ci = 0; ci < 128; ++ci)
        acc += w[od * 128 + ci] * e1[(size_t)(bb * 128 + ci) * 2048 + n];
    out[idx] = acc;
}

// ---------------------------------------------------------------------------
// Host orchestration
// ---------------------------------------------------------------------------
extern "C" void kernel_launch(void* const* d_in, const int* in_sizes, int n_in,
                              void* d_out, int out_size, void* d_ws, size_t ws_size,
                              hipStream_t stream) {
    (void)in_sizes; (void)n_in; (void)out_size; (void)ws_size;
    auto F = [&](int i) { return (const float*)d_in[i]; };
    const float* x       = F(0);
    const float* emb1    = F(1);
    const float* emb2    = F(2);
    const float* lin1_w  = F(3);  const float* lin1_b = F(4);
    const float* lin2_w  = F(5);  const float* lin2_b = F(6);
    const float* start_w = F(7);  const float* start_b = F(8);
    const float* skip0_w = F(9);  const float* skip0_b = F(10);
    const float* fw2 = F(11); const float* gw2 = F(12);
    const float* fw3 = F(13); const float* gw3 = F(14);
    const float* fw6 = F(15); const float* gw6 = F(16);
    const float* fw7 = F(17); const float* gw7 = F(18);
    const float* fb  = F(19); const float* gb  = F(20);
    const float* skw[3] = { F(21), F(22), F(23) };
    const float* skip_b = F(24);
    const float* g1_w = F(25); const float* g1_b = F(26);
    const float* g2_w = F(27); const float* g2_b = F(28);
    const float* nw[3] = { F(29), F(31), F(33) };
    const float* nb[3] = { F(30), F(32), F(34) };
    const float* skipE_w = F(35); const float* skipE_b = F(36);
    const float* end1_w = F(37);  const float* end1_b = F(38);
    const float* end2_w = F(39);  const float* end2_b = F(40);

    char* ws = (char*)d_ws;
    size_t off = 0;
    auto alloc = [&](size_t bytes) -> char* {
        off = (off + 255) & ~(size_t)255;
        char* p = ws + off;
        off += bytes;
        return p;
    };
    float* n1b  = (float*)alloc(2048ull * 40 * 4);
    float* n2b  = (float*)alloc(2048ull * 40 * 4);
    float* adj0 = (float*)alloc(2048ull * 2048 * 4);
    float* adp  = (float*)alloc(2048ull * 2048 * 4);
    float* r1   = (float*)alloc(2048ull * 4);
    float* r2   = (float*)alloc(2048ull * 4);
    __bf16* Bt1 = (__bf16*)alloc(2048ull * 2048 * 2);
    __bf16* Bt2 = (__bf16*)alloc(2048ull * 2048 * 2);
    float* hA   = (float*)alloc(16ull * 32 * 19 * 2048 * 4);
    float* hB   = (float*)alloc(16ull * 32 * 13 * 2048 * 4);
    float* hg   = (float*)alloc(16ull * 32 * 13 * 2048 * 4);
    float* g1v  = (float*)alloc(16ull * 32 * 13 * 2048 * 4);
    float* g2v  = (float*)alloc(16ull * 32 * 13 * 2048 * 4);
    __bf16* gbfA = (__bf16*)alloc(16ull * 32 * 13 * 2048 * 2);
    __bf16* gbfB = (__bf16*)alloc(16ull * 32 * 13 * 2048 * 2);
    float* mp   = (float*)alloc(16ull * 32 * 13 * 2048 * 4);
    float* skip = (float*)alloc(16ull * 64 * 2048 * 4);
    // head buffers alias the (by-then dead) adjacency scratch
    float* sbuf = adj0;                   // 8.4 MB  <= 16.8 MB
    float* e1b  = adp;                    // 16.8 MB <= 16.8 MB

    const int THR = 256;
    auto nblk = [&](long long total) { return (unsigned)((total + THR - 1) / THR); };

    // --- graph constructor ---
    k_embed<<<nblk(2 * 2048 * 40), THR, 0, stream>>>(emb1, emb2, lin1_w, lin1_b,
                                                     lin2_w, lin2_b, n1b, n2b);
    k_adj<<<dim3(128, 128), THR, 0, stream>>>(n1b, n2b, adj0);
    k_fill<<<nblk(2048 * 2048), THR, 0, stream>>>(adp, 0.0f, 2048 * 2048);
    k_topk<<<2048, THR, 0, stream>>>(adj0, adp, r1);
    k_colsum<<<nblk(2048), THR, 0, stream>>>(adp, r2);
    k_buildBt<<<dim3(64, 64), dim3(32, 32), 0, stream>>>(adp, r1, r2, Bt1, Bt2);

    // --- start conv + skip0 ---
    k_start<<<nblk(16ll * 32 * 19 * 2048), THR, 0, stream>>>(x, start_w, start_b, hA);
    k_skip0<<<nblk(16ll * 64 * 2048), THR, 0, stream>>>(x, skip0_w, skip0_b, skip);

    // --- layers ---
    float* Hcur = hA;
    float* Hnext = hB;
    int Tin = 19;
    for (int i = 0; i < 3; ++i) {
        int T = Tin - 6;                       // 13, 7, 1
        int M = 512 * T;                       // rows of mixprop GEMM
        long long elems = 16ll * 32 * T * 2048;

        k_incept<<<nblk(elems), THR, 0, stream>>>(
            Hcur, hg,
            fw2 + i * 8 * 32 * 2, fw3 + i * 8 * 32 * 3,
            fw6 + i * 8 * 32 * 6, fw7 + i * 8 * 32 * 7, fb + i * 32,
            gw2 + i * 8 * 32 * 2, gw3 + i * 8 * 32 * 3,
            gw6 + i * 8 * 32 * 6, gw7 + i * 8 * 32 * 7, gb + i * 32, T, Tin);

        k_skipconv<<<nblk(16ll * 64 * 2048), THR, 0, stream>>>(
            hg, skip, skw[i], skip_b + i * 64, T);

        // mixprop with An1 (adp)
        k_tobf<<<nblk(elems), THR, 0, stream>>>(hg, gbfA, (int)elems);
        k_gemm<<<dim3(8, M / 64), THR, 0, stream>>>(gbfA, Bt1, hg, g1v, gbfB, M, 0.05f);
        k_gemm<<<dim3(8, M / 64), THR, 0, stream>>>(gbfB, Bt1, hg, g2v,
                                                    (__bf16*)nullptr, M, 0.05f);
        k_mixconv<<<nblk(16ll * T * 2048), THR, 0, stream>>>(
            hg, g1v, g2v, g1_w + i * 32 * 96, g1_b + i * 32, mp, T, 0, nullptr, Tin);

        // mixprop with An2 (adp.T), reusing h_bf; accumulate + residual
        k_gemm<<<dim3(8, M / 64), THR, 0, stream>>>(gbfA, Bt2, hg, g1v, gbfB, M, 0.05f);
        k_gemm<<<dim3(8, M / 64), THR, 0, stream>>>(gbfB, Bt2, hg, g2v,
                                                    (__bf16*)nullptr, M, 0.05f);
        k_mixconv<<<nblk(16ll * T * 2048), THR, 0, stream>>>(
            hg, g1v, g2v, g2_w + i * 32 * 96, g2_b + i * 32, mp, T, 1, Hcur, Tin);

        k_lnorm<<<16, THR, 0, stream>>>(mp, Hnext, nw[i], nb[i], T);

        float* tmp = Hcur; Hcur = Hnext; Hnext = tmp;
        Tin = T;
    }

    // --- output head (Hcur has T=1) ---
    k_skipE<<<nblk(16ll * 64 * 2048), THR, 0, stream>>>(skip, Hcur, skipE_w, skipE_b, sbuf);
    k_end1<<<nblk(16ll * 128 * 2048), THR, 0, stream>>>(sbuf, end1_w, end1_b, e1b);
    k_end2<<<nblk(16ll * 12 * 2048), THR, 0, stream>>>(e1b, end2_w, end2_b, (float*)d_out);
}
